// CausalSelfAttention_56453050138850
// MI455X (gfx1250) — compile-verified, matrix operand order checked by reading, not run on hardware
//
#include <hip/hip_runtime.h>
#include <hip/hip_bf16.h>

// CDNA5 / gfx1250 causal self-attention: B=8, T=1024, C=1024, H=16, hs=64.
// Matmuls on v_wmma_f32_16x16x32_bf16; K/V staging via async global->LDS DMA
// (global_load_async_to_lds_b128, ASYNCcnt) with double buffering.

typedef __attribute__((ext_vector_type(16))) __bf16 v16bf;
typedef __attribute__((ext_vector_type(8)))  float  v8f;

#define LDS_PAD 8
#define KV_PAD 16   // row stride 80 bf16 = 160 B -> keeps B128 LDS dests 16B-aligned

// K-index pattern for 16-bit A/B WMMA fragments (ISA 7.12.2):
// lanes 0-15 own K chunks {0-7, 16-23}, lanes 16-31 own {8-15, 24-31}.
__device__ __forceinline__ int wmma_k_idx(int e, int hi8) {
  int v = e >> 1, h = e & 1;
  return (v < 4) ? (hi8 + 2 * v + h) : (16 + hi8 + 2 * (v - 4) + h);
}

__device__ __forceinline__ float half_reduce_max(float v) {
  v = fmaxf(v, __shfl_xor(v, 1, 32));
  v = fmaxf(v, __shfl_xor(v, 2, 32));
  v = fmaxf(v, __shfl_xor(v, 4, 32));
  v = fmaxf(v, __shfl_xor(v, 8, 32));
  return v;
}

__device__ __forceinline__ float half_reduce_sum(float v) {
  v += __shfl_xor(v, 1, 32);
  v += __shfl_xor(v, 2, 32);
  v += __shfl_xor(v, 4, 32);
  v += __shfl_xor(v, 8, 32);
  return v;
}

// Async 16-byte global -> LDS copy (VDST = LDS byte address, VADDR = 64b global).
// Generic-pointer low 32 bits are the LDS offset (ISA 10.2 aperture rules).
__device__ __forceinline__ void async_copy_b128(void* lds_ptr, const void* gptr) {
  unsigned lds_addr = (unsigned)(size_t)lds_ptr;
  asm volatile("global_load_async_to_lds_b128 %0, %1, off"
               :: "v"(lds_addr), "v"(gptr) : "memory");
}

// ---------------------------------------------------------------------------
// Tiled GEMM:  C[M,N] = A[M,K] * W[K,N] + bias[N]
// Block: 256 threads = 8 wave32 in a 2(M) x 4(N) grid. Block tile 64M x 128N.
// Each wave register-blocks a 32x32 output tile: 4 WMMAs per 32-wide K step.
// ---------------------------------------------------------------------------
template <bool A_BF16, bool OUT_BF16>
__global__ __launch_bounds__(256)
void gemm_bias_wmma(const void* __restrict__ Av, const float* __restrict__ W,
                    const float* __restrict__ bias, void* __restrict__ Cout,
                    int M, int N, int K)
{
  __shared__ __bf16 As[64][32 + LDS_PAD];
  __shared__ __bf16 Bs[32][128 + LDS_PAD];

  const int tid  = threadIdx.x;
  const int lane = tid & 31;
  const int wave = tid >> 5;
  const int wy   = wave >> 2;          // 0..1 : 32-row slice
  const int wx   = wave & 3;           // 0..3 : 32-col slice
  const int m0   = blockIdx.y * 64;
  const int nblk = blockIdx.x * 128;
  const int hi8  = (lane >> 4) << 3;
  const int l16  = lane & 15;

  v8f acc[2][2] = {};

  for (int kt = 0; kt < K; kt += 32) {
    __syncthreads();
    // Stage A tile (64x32), convert f32->bf16 if needed.
    for (int i = tid; i < 64 * 32; i += 256) {
      int r = i >> 5, c = i & 31;
      size_t src = (size_t)(m0 + r) * K + kt + c;
      As[r][c] = A_BF16 ? ((const __bf16*)Av)[src]
                        : (__bf16)(((const float*)Av)[src]);
    }
    // Stage B tile (32x128).
    for (int i = tid; i < 32 * 128; i += 256) {
      int r = i >> 7, c = i & 127;
      Bs[r][c] = (__bf16)W[(size_t)(kt + r) * N + nblk + c];
    }
    // Prefetch next K tiles into cache while WMMAs run.
    if (kt + 32 < K) {
      __builtin_prefetch(&W[(size_t)(kt + 32 + (tid >> 3)) * N + nblk + (tid & 7) * 16], 0, 3);
      if (A_BF16)
        __builtin_prefetch(&((const __bf16*)Av)[(size_t)(m0 + (tid >> 2)) * K + kt + 32], 0, 3);
      else
        __builtin_prefetch(&((const float*)Av)[(size_t)(m0 + (tid >> 2)) * K + kt + 32], 0, 3);
    }
    __syncthreads();

    v16bf af[2], bf[2];
#pragma unroll
    for (int e = 0; e < 16; ++e) {
      int k = wmma_k_idx(e, hi8);
      af[0][e] = As[wy * 32 + l16][k];
      af[1][e] = As[wy * 32 + 16 + l16][k];
      bf[0][e] = Bs[k][wx * 32 + l16];
      bf[1][e] = Bs[k][wx * 32 + 16 + l16];
    }
#pragma unroll
    for (int i = 0; i < 2; ++i)
#pragma unroll
      for (int j = 0; j < 2; ++j)
        acc[i][j] = __builtin_amdgcn_wmma_f32_16x16x32_bf16(
            false, af[i], false, bf[j], (short)0, acc[i][j], false, false);
  }

#pragma unroll
  for (int i = 0; i < 2; ++i) {
#pragma unroll
    for (int j = 0; j < 2; ++j) {
      const int ncol = nblk + wx * 32 + j * 16 + l16;
      const float bv = bias[ncol];
#pragma unroll
      for (int r = 0; r < 8; ++r) {
        int row = m0 + wy * 32 + i * 16 + r + hi8;
        float v = acc[i][j][r] + bv;
        if (OUT_BF16)
          ((__bf16*)Cout)[(size_t)row * N + ncol] = (__bf16)v;
        else
          ((float*)Cout)[(size_t)row * N + ncol] = v;
      }
    }
  }
}

// ---------------------------------------------------------------------------
// Flash attention: grid (T/128, B*H). Block = 8 waves; wave w owns q rows
// [blockIdx.x*128 + 16w, +16). K/V tiles stream global->LDS via async DMA with
// double buffering; WMMA for S = Q K^T and O += P V; online softmax.
// qkv is bf16 [B*T, 3C]; out is bf16 [B*T, C] with col = h*64 + d.
// ---------------------------------------------------------------------------
__global__ __launch_bounds__(256)
void attn_flash_wmma(const __bf16* __restrict__ qkv, __bf16* __restrict__ outb)
{
  const int T = 1024, C = 1024, C3 = 3072, HS = 64;
  __shared__ __bf16 Ks[2][32][HS + KV_PAD];
  __shared__ __bf16 Vs[2][32][HS + KV_PAD];
  __shared__ __bf16 Ps[8][16][32];          // per-wave P transpose buffer

  const int tid  = threadIdx.x;
  const int lane = tid & 31;
  const int wave = tid >> 5;
  const int b    = blockIdx.y >> 4;
  const int h    = blockIdx.y & 15;
  const int qr0  = blockIdx.x * 128 + wave * 16;
  const int hi8  = (lane >> 4) << 3;
  const int l16  = lane & 15;
  const float scale = 0.125f;               // 1/sqrt(64)

  // Per-thread async-copy slice: 16 bytes of one K row and one V row per tile.
  const int cr = tid >> 3;                  // 0..31 : key row in tile
  const int cc = (tid & 7) * 8;             // 0,8,..,56 : d offset (8 bf16 = 16B)

  // Q fragments for both 32-wide d chunks, resident in registers.
  v16bf qf0, qf1;
  {
    size_t qbase = (size_t)(b * T + qr0 + l16) * C3 + h * HS;
#pragma unroll
    for (int e = 0; e < 16; ++e) {
      int k = wmma_k_idx(e, hi8);
      qf0[e] = qkv[qbase + k];
      qf1[e] = qkv[qbase + 32 + k];
    }
  }

  v8f o[4] = {{}, {}, {}, {}};
  float mprev[8], lsum[8];
#pragma unroll
  for (int r = 0; r < 8; ++r) { mprev[r] = -1e30f; lsum[r] = 0.f; }

  const int nk = blockIdx.x * 4 + 4;        // 32-key tiles covering keys <= block max q

  // Prologue: kick off async staging of tile 0 into buffer 0.
  {
    size_t base = (size_t)(b * T + cr) * C3 + h * HS + cc;
    async_copy_b128(&Ks[0][cr][cc], qkv + base + C);
    async_copy_b128(&Vs[0][cr][cc], qkv + base + 2 * C);
  }

  for (int j = 0; j < nk; ++j) {
    const int k0  = j * 32;
    const int buf = j & 1;

    // Issue async staging of tile j+1 into the other buffer, then wait for
    // this wave's tile-j copies (2 older ops) and barrier so all waves' data
    // is resident before compute.
    if (j + 1 < nk) {
      size_t base = (size_t)(b * T + (j + 1) * 32 + cr) * C3 + h * HS + cc;
      async_copy_b128(&Ks[buf ^ 1][cr][cc], qkv + base + C);
      async_copy_b128(&Vs[buf ^ 1][cr][cc], qkv + base + 2 * C);
      asm volatile("s_wait_asynccnt 0x2" ::: "memory");
    } else {
      asm volatile("s_wait_asynccnt 0x0" ::: "memory");
    }
    __syncthreads();

    // S = Q K^T for two 16-key subtiles (2 WMMAs each over hs=64).
    v8f s0 = {}, s1 = {};
    {
      v16bf kf00, kf01, kf10, kf11;
#pragma unroll
      for (int e = 0; e < 16; ++e) {
        int k = wmma_k_idx(e, hi8);
        kf00[e] = Ks[buf][l16][k];
        kf01[e] = Ks[buf][l16][32 + k];
        kf10[e] = Ks[buf][16 + l16][k];
        kf11[e] = Ks[buf][16 + l16][32 + k];
      }
      s0 = __builtin_amdgcn_wmma_f32_16x16x32_bf16(false, qf0, false, kf00, (short)0, s0, false, false);
      s0 = __builtin_amdgcn_wmma_f32_16x16x32_bf16(false, qf1, false, kf01, (short)0, s0, false, false);
      s1 = __builtin_amdgcn_wmma_f32_16x16x32_bf16(false, qf0, false, kf10, (short)0, s1, false, false);
      s1 = __builtin_amdgcn_wmma_f32_16x16x32_bf16(false, qf1, false, kf11, (short)0, s1, false, false);
    }

    // Online softmax update per row (rows r+hi8; cols spread over 16-lane half).
#pragma unroll
    for (int r = 0; r < 8; ++r) {
      int qg = qr0 + r + hi8;
      float a0 = s0[r] * scale; if (k0 + l16      > qg) a0 = -1e30f;
      float a1 = s1[r] * scale; if (k0 + 16 + l16 > qg) a1 = -1e30f;
      float rmax = half_reduce_max(fmaxf(a0, a1));
      float mnew = fmaxf(mprev[r], rmax);
      float alpha = __expf(mprev[r] - mnew);
      float p0 = __expf(a0 - mnew);
      float p1 = __expf(a1 - mnew);
      float rs = half_reduce_sum(p0 + p1);
      lsum[r] = lsum[r] * alpha + rs;
      mprev[r] = mnew;
#pragma unroll
      for (int dc = 0; dc < 4; ++dc) o[dc][r] *= alpha;
      // Stash P in LDS for C-layout -> A-layout transpose.
      Ps[wave][r + hi8][l16]      = (__bf16)p0;
      Ps[wave][r + hi8][16 + l16] = (__bf16)p1;
    }
    asm volatile("s_wait_dscnt 0" ::: "memory");
    __builtin_amdgcn_wave_barrier();

    v16bf pf;
#pragma unroll
    for (int e = 0; e < 16; ++e) pf[e] = Ps[wave][l16][wmma_k_idx(e, hi8)];

    // O += P V over 32 keys, for 4 d-chunks of 16.
#pragma unroll
    for (int dc = 0; dc < 4; ++dc) {
      v16bf vf;
#pragma unroll
      for (int e = 0; e < 16; ++e) vf[e] = Vs[buf][wmma_k_idx(e, hi8)][dc * 16 + l16];
      o[dc] = __builtin_amdgcn_wmma_f32_16x16x32_bf16(false, pf, false, vf,
                                                      (short)0, o[dc], false, false);
    }

    // All waves must finish reading buffer `buf` before iteration j+1 issues
    // async writes into buffer `buf` (the j+2 tile target is buf of j).
    __syncthreads();
  }

  // Normalize and write out as bf16 [B*T, C] with col = h*64 + d.
#pragma unroll
  for (int r = 0; r < 8; ++r) {
    float inv = 1.0f / lsum[r];
    size_t row = (size_t)(b * T + qr0 + r + hi8) * C + h * HS;
#pragma unroll
    for (int dc = 0; dc < 4; ++dc)
      outb[row + dc * 16 + l16] = (__bf16)(o[dc][r] * inv);
  }
}

// ---------------------------------------------------------------------------
extern "C" void kernel_launch(void* const* d_in, const int* in_sizes, int n_in,
                              void* d_out, int out_size, void* d_ws, size_t ws_size,
                              hipStream_t stream) {
  const float* x     = (const float*)d_in[0];   // [8,1024,1024]
  const float* Wqkv  = (const float*)d_in[1];   // [1024,3072]
  const float* bqkv  = (const float*)d_in[2];   // [3072]
  const float* Wproj = (const float*)d_in[3];   // [1024,1024]
  const float* bproj = (const float*)d_in[4];   // [1024]
  float* out = (float*)d_out;                   // [8,1024,1024]

  const int M = 8 * 1024, C = 1024, C3 = 3072;
  __bf16* qkvb  = (__bf16*)d_ws;                       // 48 MB
  __bf16* attnb = qkvb + (size_t)M * C3;               // 16 MB

  dim3 blk(256);
  // qkv = x @ W_qkv + b_qkv  (bf16 out)
  gemm_bias_wmma<false, true><<<dim3(C3 / 128, M / 64), blk, 0, stream>>>(
      x, Wqkv, bqkv, qkvb, M, C3, C);
  // flash attention
  attn_flash_wmma<<<dim3(1024 / 128, 8 * 16), blk, 0, stream>>>(qkvb, attnb);
  // out = attn @ W_proj + b_proj  (f32 out)
  gemm_bias_wmma<true, false><<<dim3(C / 128, M / 64), blk, 0, stream>>>(
      attnb, Wproj, bproj, out, M, C, C);
}
